// RelationNetworks_54150947668288
// MI455X (gfx1250) — matrix operand order repaired and checked by reading, not compiled, not run on hardware
//
#include <hip/hip_runtime.h>
#include <hip/hip_bf16.h>

typedef __attribute__((ext_vector_type(16))) _Float16 v16h;
typedef __attribute__((ext_vector_type(8)))  _Float16 v8h;
typedef __attribute__((ext_vector_type(8)))  float    v8f;

#define BATCH 64
#define SEQ 30
#define CH 24
#define HID 256
#define ASTRIDE 264   // 256 + 8 halfs pad; multiple of 8 keeps 16B alignment

// ---------------- conv 3x3 stride2 pad1 ----------------
__global__ void conv3x3s2(const float* __restrict__ x, const float* __restrict__ w,
                          float* __restrict__ y, int Cin, int Hin, int Cout, int Ho) {
  int idx = blockIdx.x * blockDim.x + threadIdx.x;
  int total = BATCH * Cout * Ho * Ho;
  if (idx >= total) return;
  int wo = idx % Ho; int t = idx / Ho;
  int ho = t % Ho; t /= Ho;
  int co = t % Cout; int b = t / Cout;
  const float* wp = w + co * Cin * 9;
  float acc = 0.f;
  for (int ci = 0; ci < Cin; ++ci) {
    const float* xp = x + ((size_t)(b * Cin + ci)) * Hin * Hin;
    for (int kh = 0; kh < 3; ++kh) {
      int hi = ho * 2 - 1 + kh;
      if ((unsigned)hi >= (unsigned)Hin) continue;
      for (int kw = 0; kw < 3; ++kw) {
        int wi = wo * 2 - 1 + kw;
        if ((unsigned)wi >= (unsigned)Hin) continue;
        acc += xp[hi * Hin + wi] * wp[(ci * 3 + kh) * 3 + kw];
      }
    }
  }
  y[idx] = acc;
}

// ---------------- batchnorm stats (one block per channel) ----------------
__global__ void bnstats(const float* __restrict__ y, float* __restrict__ stats,
                        int C, int HW) {
  int c = blockIdx.x;
  int N = BATCH * HW;
  float s = 0.f, ss = 0.f;
  for (int i = threadIdx.x; i < N; i += 256) {
    int b = i / HW, p = i % HW;
    float v = y[((size_t)(b * C + c)) * HW + p];
    s += v; ss += v * v;
  }
  __shared__ float sh[256], sh2[256];
  sh[threadIdx.x] = s; sh2[threadIdx.x] = ss;
  __syncthreads();
  for (int o = 128; o > 0; o >>= 1) {
    if (threadIdx.x < (unsigned)o) { sh[threadIdx.x] += sh[threadIdx.x + o]; sh2[threadIdx.x] += sh2[threadIdx.x + o]; }
    __syncthreads();
  }
  if (threadIdx.x == 0) {
    float m = sh[0] / (float)N;
    float var = sh2[0] / (float)N - m * m;
    stats[c] = m;
    stats[C + c] = rsqrtf(var + 1e-5f);
  }
}

__global__ void bnrelu(float* __restrict__ y, const float* __restrict__ stats,
                       const float* __restrict__ gamma, const float* __restrict__ beta,
                       int C, int HW) {
  int idx = blockIdx.x * blockDim.x + threadIdx.x;
  int total = BATCH * C * HW;
  if (idx >= total) return;
  int c = (idx / HW) % C;
  float v = (y[idx] - stats[c]) * stats[C + c] * gamma[c] + beta[c];
  y[idx] = v > 0.f ? v : 0.f;
}

// ---------------- embed + LSTM (one block per batch row) ----------------
__global__ void lstm_kernel(const int* __restrict__ q, const float* __restrict__ emb,
                            const float* __restrict__ wih, const float* __restrict__ whh,
                            const float* __restrict__ bias, float* __restrict__ hout) {
  int b = blockIdx.x;
  int t = threadIdx.x;          // 128 threads, one per hidden unit
  __shared__ float h[128], c[128], xt[32];
  h[t] = 0.f; c[t] = 0.f;
  __syncthreads();
  for (int s = 0; s < SEQ; ++s) {
    if (t < 32) xt[t] = emb[q[b * SEQ + s] * 32 + t];
    __syncthreads();
    float a0 = bias[t], a1 = bias[128 + t], a2 = bias[256 + t], a3 = bias[384 + t];
    for (int k = 0; k < 32; ++k) {
      float xv = xt[k];
      a0 += wih[t * 32 + k] * xv;
      a1 += wih[(128 + t) * 32 + k] * xv;
      a2 += wih[(256 + t) * 32 + k] * xv;
      a3 += wih[(384 + t) * 32 + k] * xv;
    }
    for (int k = 0; k < 128; ++k) {
      float hv = h[k];
      a0 += whh[t * 128 + k] * hv;
      a1 += whh[(128 + t) * 128 + k] * hv;
      a2 += whh[(256 + t) * 128 + k] * hv;
      a3 += whh[(384 + t) * 128 + k] * hv;
    }
    __syncthreads();
    float ig = 1.f / (1.f + __expf(-a0));
    float fg = 1.f / (1.f + __expf(-a1));
    float gg = tanhf(a2);
    float og = 1.f / (1.f + __expf(-a3));
    float cn = fg * c[t] + ig * gg;
    c[t] = cn;
    h[t] = og * tanhf(cn);
    __syncthreads();
  }
  hout[b * 128 + t] = h[t];
}

// ---------------- pa / pb : (B*64) x 26 -> 256 ----------------
__global__ void pab_kernel(const float* __restrict__ x4, const float* __restrict__ w1,
                           float* __restrict__ out, int coff) {
  int idx = blockIdx.x * blockDim.x + threadIdx.x;   // B*64*256
  int n = idx & 255; int t = idx >> 8;
  int p = t & 63; int b = t >> 6;
  int hh = p >> 3, ww = p & 7;
  const float* wp = w1 + n * 180 + coff;
  float acc = 0.f;
  for (int c = 0; c < CH; ++c)
    acc += x4[((size_t)(b * CH + c)) * 64 + p] * wp[c];
  float cx = -4.f + 8.f * (float)ww / 7.f;
  float cy = -4.f + 8.f * (float)hh / 7.f;
  acc += cx * wp[24] + cy * wp[25];
  out[idx] = acc;
}

// ---------------- pq : 64 x 128 -> 256 ----------------
__global__ void pq_kernel(const float* __restrict__ h, const float* __restrict__ w1,
                          float* __restrict__ out) {
  int idx = blockIdx.x * blockDim.x + threadIdx.x;   // 64*256
  int n = idx & 255; int b = idx >> 8;
  const float* wp = w1 + n * 180 + 52;
  float acc = 0.f;
  for (int k = 0; k < 128; ++k) acc += h[b * 128 + k] * wp[k];
  out[idx] = acc;
}

// ---------------- weight swizzle into WMMA B-fragment order ----------------
// frag index = ((nt*8 + kt)*32 + lane)*16 + e ; value = W[n, k]
// n = nt*16 + (lane&15), k = kt*32 + (lane>>4)*16 + e
__global__ void wswz_kernel(const float* __restrict__ w2, const float* __restrict__ w3,
                            const float* __restrict__ w4, _Float16* __restrict__ out) {
  int idx = blockIdx.x * blockDim.x + threadIdx.x;   // 3*65536
  int l = idx >> 16; int r = idx & 65535;
  int e = r & 15, lane = (r >> 4) & 31, kt = (r >> 9) & 7, nt = r >> 12;
  int ln = lane & 15, half = lane >> 4;
  int n = nt * 16 + ln;
  int k = kt * 32 + half * 16 + e;
  const float* W = (l == 0) ? w2 : (l == 1) ? w3 : w4;
  out[idx] = (_Float16)W[n * 256 + k];
}

__global__ void zero_kernel(float* __restrict__ p, int n) {
  int idx = blockIdx.x * blockDim.x + threadIdx.x;
  if (idx < n) p[idx] = 0.f;
}

// ---------------- main relational g-MLP with WMMA ----------------
__global__ void __launch_bounds__(256)
rn_g_kernel(const float* __restrict__ pa, const float* __restrict__ pb,
            const float* __restrict__ pq, const float* __restrict__ b1,
            const _Float16* __restrict__ wfrag,
            const float* __restrict__ b2, const float* __restrict__ b3,
            const float* __restrict__ b4, float* __restrict__ g_sum) {
  __shared__ __align__(16) _Float16 Abuf[64 * ASTRIDE];
  int b = blockIdx.x >> 6;
  int t = blockIdx.x & 63;
  int i0 = (t & 7) * 8;
  int j0 = (t >> 3) * 8;
  int tid = threadIdx.x;

  // ---- build g1 tile: relu(pa[i] + pb[j] + pq + b1) -> f16 LDS
  float base = pq[b * 256 + tid] + b1[tid];
  float pav[8], pbv[8];
  for (int r = 0; r < 8; ++r) {
    pav[r] = pa[((size_t)(b * 64 + i0 + r)) * 256 + tid];
    pbv[r] = pb[((size_t)(b * 64 + j0 + r)) * 256 + tid];
  }
  for (int r = 0; r < 64; ++r) {
    float v = pav[r & 7] + pbv[r >> 3] + base;
    Abuf[r * ASTRIDE + tid] = (_Float16)(v > 0.f ? v : 0.f);
  }
  __syncthreads();

  int lane = tid & 31, wave = tid >> 5;
  int ln = lane & 15, half = lane >> 4;
  int nt0 = wave * 2;
  const float* biases[3] = {b2, b3, b4};

  for (int l = 0; l < 3; ++l) {
    v8f acc[4][2];
    for (int mt = 0; mt < 4; ++mt)
      for (int j = 0; j < 2; ++j) { v8f z = {}; acc[mt][j] = z; }

    const _Float16* wl = wfrag + l * 65536;
    for (int kt = 0; kt < 8; ++kt) {
      v16h afrag[4];
      for (int mt = 0; mt < 4; ++mt) {
        const _Float16* ap = &Abuf[(mt * 16 + ln) * ASTRIDE + kt * 32 + half * 16];
        v8h lo = *(const v8h*)ap;
        v8h hi = *(const v8h*)(ap + 8);
        afrag[mt] = __builtin_shufflevector(lo, hi, 0,1,2,3,4,5,6,7,8,9,10,11,12,13,14,15);
      }
      for (int j = 0; j < 2; ++j) {
        const _Float16* wp = wl + (((nt0 + j) * 8 + kt) * 32 + lane) * 16;
        v8h lo = *(const v8h*)wp;
        v8h hi = *(const v8h*)(wp + 8);
        v16h bfrag = __builtin_shufflevector(lo, hi, 0,1,2,3,4,5,6,7,8,9,10,11,12,13,14,15);
        for (int mt = 0; mt < 4; ++mt)
          acc[mt][j] = __builtin_amdgcn_wmma_f32_16x16x32_f16(
              false, afrag[mt], false, bfrag, (short)0, acc[mt][j], false, false);
      }
    }
    __syncthreads();   // all reads of Abuf done before in-place overwrite

    if (l < 2) {
      for (int j = 0; j < 2; ++j) {
        int n = (nt0 + j) * 16 + ln;
        float bv = biases[l][n];
        for (int mt = 0; mt < 4; ++mt)
          for (int r = 0; r < 8; ++r) {
            float v = acc[mt][j][r] + bv;
            Abuf[(mt * 16 + half * 8 + r) * ASTRIDE + n] = (_Float16)(v > 0.f ? v : 0.f);
          }
      }
      __syncthreads();
    } else {
      for (int j = 0; j < 2; ++j) {
        int n = (nt0 + j) * 16 + ln;
        float bv = biases[2][n];
        float s = 0.f;
        for (int mt = 0; mt < 4; ++mt)
          for (int r = 0; r < 8; ++r) {
            float v = acc[mt][j][r] + bv;
            s += (v > 0.f ? v : 0.f);
          }
        s += __shfl_xor(s, 16, 32);   // combine M halves (same column n)
        if (half == 0) atomicAdd(&g_sum[b * 256 + n], s);
      }
    }
  }
}

// ---------------- f layers ----------------
__global__ void fhid_kernel(const float* __restrict__ gsum, const float* __restrict__ w,
                            const float* __restrict__ bias, float* __restrict__ out) {
  int idx = blockIdx.x * blockDim.x + threadIdx.x;   // 64*256
  int n = idx & 255; int b = idx >> 8;
  float acc = bias[n];
  for (int k = 0; k < 256; ++k) acc += gsum[b * 256 + k] * w[n * 256 + k];
  out[idx] = acc > 0.f ? acc : 0.f;
}

__global__ void fout_kernel(const float* __restrict__ hid, const float* __restrict__ w,
                            const float* __restrict__ bias, float* __restrict__ out) {
  int idx = blockIdx.x * blockDim.x + threadIdx.x;   // 64*29
  if (idx >= BATCH * 29) return;
  int n = idx % 29; int b = idx / 29;
  float acc = bias[n];
  for (int k = 0; k < 256; ++k) acc += hid[b * 256 + k] * w[n * 256 + k];
  out[idx] = acc;
}

extern "C" void kernel_launch(void* const* d_in, const int* in_sizes, int n_in,
                              void* d_out, int out_size, void* d_ws, size_t ws_size,
                              hipStream_t stream) {
  const float* image = (const float*)d_in[0];
  const int*   question = (const int*)d_in[1];
  const float *convw[4], *convg[4], *convb[4];
  const float *embed, *wih, *whh, *lstmb;
  const float *gw[4], *gb[4], *f0w, *f0b, *f1w, *f1b;

  if (in_sizes[2] == 648) {
    // insertion-order flattening: convs{w,gamma,beta}..., embed, lstm{w_ih,w_hh,b}, g{w,b}..., f{w,b}...
    for (int l = 0; l < 4; ++l) {
      convw[l] = (const float*)d_in[2 + 3 * l];
      convg[l] = (const float*)d_in[3 + 3 * l];
      convb[l] = (const float*)d_in[4 + 3 * l];
    }
    embed = (const float*)d_in[14];
    wih = (const float*)d_in[15]; whh = (const float*)d_in[16]; lstmb = (const float*)d_in[17];
    for (int l = 0; l < 4; ++l) { gw[l] = (const float*)d_in[18 + 2 * l]; gb[l] = (const float*)d_in[19 + 2 * l]; }
    f0w = (const float*)d_in[26]; f0b = (const float*)d_in[27];
    f1w = (const float*)d_in[28]; f1b = (const float*)d_in[29];
  } else {
    // sorted-pytree flattening: convs{beta,gamma,w}..., embed, f{b,w}..., g{b,w}..., lstm{b,w_hh,w_ih}
    for (int l = 0; l < 4; ++l) {
      convb[l] = (const float*)d_in[2 + 3 * l];
      convg[l] = (const float*)d_in[3 + 3 * l];
      convw[l] = (const float*)d_in[4 + 3 * l];
    }
    embed = (const float*)d_in[14];
    f0b = (const float*)d_in[15]; f0w = (const float*)d_in[16];
    f1b = (const float*)d_in[17]; f1w = (const float*)d_in[18];
    for (int l = 0; l < 4; ++l) { gb[l] = (const float*)d_in[19 + 2 * l]; gw[l] = (const float*)d_in[20 + 2 * l]; }
    lstmb = (const float*)d_in[27]; whh = (const float*)d_in[28]; wih = (const float*)d_in[29];
  }

  char* ws = (char*)d_ws;
  float*    c1     = (float*)(ws + 0);          // 64*24*64*64*4 = 25165824
  float*    c2     = (float*)(ws + 25165824);   // 6291456
  float*    c3     = (float*)(ws + 31457280);   // 1572864
  float*    c4     = (float*)(ws + 33030144);   // 393216
  float*    stats  = (float*)(ws + 33423360);   // 4*64 floats
  float*    hout   = (float*)(ws + 33424384);   // 32768
  float*    pa     = (float*)(ws + 33457152);   // 4194304
  float*    pb     = (float*)(ws + 37651456);   // 4194304
  float*    pq     = (float*)(ws + 41845760);   // 65536
  _Float16* wfrag  = (_Float16*)(ws + 41911296);// 393216
  float*    gsum   = (float*)(ws + 42304512);   // 65536
  float*    fhid   = (float*)(ws + 42370048);   // 65536

  // conv stack
  const int Hin[4]  = {128, 64, 32, 16};
  const int Ho[4]   = {64, 32, 16, 8};
  const int Cin[4]  = {3, CH, CH, CH};
  float* cout[4] = {c1, c2, c3, c4};
  const float* cin = image;
  for (int l = 0; l < 4; ++l) {
    int total = BATCH * CH * Ho[l] * Ho[l];
    conv3x3s2<<<(total + 255) / 256, 256, 0, stream>>>(cin, convw[l], cout[l], Cin[l], Hin[l], CH, Ho[l]);
    bnstats<<<CH, 256, 0, stream>>>(cout[l], stats + 64 * l, CH, Ho[l] * Ho[l]);
    bnrelu<<<(total + 255) / 256, 256, 0, stream>>>(cout[l], stats + 64 * l, convg[l], convb[l], CH, Ho[l] * Ho[l]);
    cin = cout[l];
  }

  // question path
  lstm_kernel<<<BATCH, 128, 0, stream>>>(question, embed, wih, whh, lstmb, hout);

  // decomposed g1 projections
  pab_kernel<<<4096, 256, 0, stream>>>(c4, gw[0], pa, 0);
  pab_kernel<<<4096, 256, 0, stream>>>(c4, gw[0], pb, 26);
  pq_kernel<<<64, 256, 0, stream>>>(hout, gw[0], pq);

  // weight fragments for g2..g4
  wswz_kernel<<<768, 256, 0, stream>>>(gw[1], gw[2], gw[3], wfrag);
  zero_kernel<<<64, 256, 0, stream>>>(gsum, BATCH * 256);

  // relational core (WMMA)
  rn_g_kernel<<<4096, 256, 0, stream>>>(pa, pb, pq, gb[0], wfrag, gb[1], gb[2], gb[3], gsum);

  // f layers
  fhid_kernel<<<64, 256, 0, stream>>>(gsum, f0w, f0b, fhid);
  fout_kernel<<<8, 256, 0, stream>>>(fhid, f1w, f1b, (float*)d_out);
}
